// PointMambaLayer_81071802679641
// MI455X (gfx1250) — matrix operand order repaired and checked by reading, compiled, not run on hardware
//
#include <hip/hip_runtime.h>

// ---------------------------------------------------------------------------
// PointMamba (Mamba2/SSD) forward for MI455X (gfx1250), wave32 + WMMA bf16.
// All matrix contractions lower to v_wmma_f32_16x16x32_bf16; all fragment
// loads are contiguous (feature-major transposed copies for K-major operands);
// k_ydiag overlaps an async global->LDS tile copy with VALU work.
// ---------------------------------------------------------------------------

typedef __attribute__((ext_vector_type(16))) __bf16 v16bf;
typedef __attribute__((ext_vector_type(8)))  float  v8f;
typedef unsigned short u16;
typedef unsigned int   u32;
typedef unsigned long long u64;

constexpr int kN      = 16384;  // tokens
constexpr int kC      = 128;    // model dim
constexpr int kDIN    = 256;    // d_inner
constexpr int kNS     = 256;    // d_state
constexpr int kH      = 16;     // heads
constexpr int kP      = 16;     // headdim
constexpr int kDPROJ  = 1040;   // z(256) + xBC(768) + dt(16)
constexpr int kCONV   = 768;
constexpr int kCHUNK  = 64;
constexpr int kNCH    = 256;    // number of chunks

union BFrag { v16bf v; u16 u[16]; };

__device__ __forceinline__ u16 f2bf(float f) {
  u32 u = __float_as_uint(f);
  u32 r = u + 0x7FFFu + ((u >> 16) & 1u);   // round-to-nearest-even
  return (u16)(r >> 16);
}
__device__ __forceinline__ float bf2f(u16 h) {
  return __uint_as_float(((u32)h) << 16);
}
__device__ __forceinline__ float sigmoidf_(float x) { return 1.f / (1.f + __expf(-x)); }
__device__ __forceinline__ float siluf_(float x)    { return x * sigmoidf_(x); }
__device__ __forceinline__ float softplusf_(float x){ return (x > 20.f) ? x : log1pf(__expf(x)); }

__device__ __forceinline__ v8f wmma_bf16(v16bf a, v16bf b, v8f c) {
  // D = A(16x32 bf16) x B(32x16 bf16) + C(16x16 f32)
  return __builtin_amdgcn_wmma_f32_16x16x32_bf16(false, a, false, b, (short)0, c, false, false);
}

// A fragment loader, row-major source: A[row][k], lda in elements.
// Lane layout (CDNA5 ISA 7.12.2): row = lane&15; half i -> K = k0 + hi*8 + (i<8 ? i : i+8)
__device__ __forceinline__ v16bf load_a(const u16* src, int lda, int row, int k0, int hi) {
  BFrag a;
  const u16* p = src + row * lda + k0 + hi * 8;
#pragma unroll
  for (int i = 0; i < 8; ++i) { a.u[i] = p[i]; a.u[8 + i] = p[16 + i]; }
  return a.v;
}
// B fragment: col = lane&15; half i -> K = k0 + hi*8 + (i<8 ? i : i+8).  Source must be
// column-major over K (K contiguous per column): src[col*nst + k].
__device__ __forceinline__ v16bf load_b(const u16* src, int nst, int col, int k0, int hi) {
  BFrag b;
  const u16* p = src + col * nst + k0 + hi * 8;
#pragma unroll
  for (int i = 0; i < 8; ++i) { b.u[i] = p[i]; b.u[8 + i] = p[16 + i]; }
  return b.v;
}

// ---------------------------------------------------------------------------
// 0) hidden = f + f_gs @ pos_embed             <<<N, 128>>>
__global__ void k_embed(const float* __restrict__ f, const float* __restrict__ fgs,
                        const float* __restrict__ pos, float* __restrict__ hidden) {
  int t = blockIdx.x, c = threadIdx.x;
  float acc = f[t * kC + c];
#pragma unroll
  for (int g = 0; g < 8; ++g) acc += fgs[t * 8 + g] * pos[g * kC + c];
  hidden[t * kC + c] = acc;
}

// 1) residual = hidden (+ residual); xn = rmsnorm(residual)*ln_w -> bf16   <<<N,128>>>
__global__ void k_prep_rms(const float* __restrict__ hidden, float* __restrict__ residual,
                           const float* __restrict__ lnw, u16* __restrict__ xnbf, int first) {
  __shared__ float s[kC];
  int t = blockIdx.x, c = threadIdx.x;
  float r = hidden[t * kC + c] + (first ? 0.f : residual[t * kC + c]);
  residual[t * kC + c] = r;
  s[c] = r * r;
  __syncthreads();
  for (int o = kC / 2; o > 0; o >>= 1) { if (c < o) s[c] += s[c + o]; __syncthreads(); }
  float inv = __frsqrt_rn(s[0] * (1.f / kC) + 1e-5f);
  xnbf[t * kC + c] = f2bf(r * inv * lnw[c]);
}

// 2) W_in^T -> bf16 [DPROJ][C]     <<<DPROJ*C/256, 256>>>
__global__ void k_wt_in(const float* __restrict__ Wi, u16* __restrict__ Wt) {
  int idx = blockIdx.x * 256 + threadIdx.x;       // idx = n*C + k
  int n = idx / kC, k = idx % kC;
  Wt[idx] = f2bf(Wi[k * kDPROJ + n]);
}
// 3) W_out^T -> bf16 [C][DIN]      <<<C*DIN/256, 256>>>
__global__ void k_wt_out(const float* __restrict__ Wo, u16* __restrict__ Wt) {
  int idx = blockIdx.x * 256 + threadIdx.x;       // idx = c*DIN + d
  int c = idx / kDIN, d = idx % kDIN;
  Wt[idx] = f2bf(Wo[d * kC + c]);
}

// 4) zxbcdt[N,1040] = xn_bf16 @ W_in  (M=16384,K=128,Nout=1040)  <<<8320, 256>>>
__global__ void k_inproj(const u16* __restrict__ xn, const u16* __restrict__ Wt,
                         float* __restrict__ zx) {
  int wid = threadIdx.x >> 5, lane = threadIdx.x & 31;
  int lo = lane & 15, hi = lane >> 4;
  int tile = blockIdx.x * 8 + wid;                 // 1024 * 65 tiles
  int mt = tile / 65, nt = tile % 65;
  __builtin_prefetch(Wt + (nt * 16 + lo) * kC, 0, 1);   // global_prefetch_b8
  v8f acc = {};
#pragma unroll
  for (int kk = 0; kk < kC; kk += 32) {
    v16bf a = load_a(xn, kC, mt * 16 + lo, kk, hi);
    v16bf b = load_b(Wt, kC, nt * 16 + lo, kk, hi);
    acc = wmma_bf16(a, b, acc);
  }
#pragma unroll
  for (int v = 0; v < 8; ++v)
    zx[(mt * 16 + v + hi * 8) * kDPROJ + nt * 16 + lo] = acc[v];
}

// 5) causal depthwise conv4 + SiLU over xBC; split -> x/B/C bf16   <<<N*768/256,256>>>
__global__ void k_conv(const float* __restrict__ zx, const float* __restrict__ cw,
                       const float* __restrict__ cb, u16* __restrict__ xbf,
                       u16* __restrict__ Bbf, u16* __restrict__ Cbf) {
  int idx = blockIdx.x * 256 + threadIdx.x;
  int t = idx / kCONV, ch = idx % kCONV;
  float acc = cb[ch];
#pragma unroll
  for (int k = 0; k < 4; ++k) {
    int tt = t + k - 3;
    if (tt >= 0) acc += zx[tt * kDPROJ + kDIN + ch] * cw[ch * 4 + k];
  }
  u16 val = f2bf(siluf_(acc));
  if (ch < 256)       xbf[t * 256 + ch]       = val;
  else if (ch < 512)  Bbf[t * 256 + ch - 256] = val;
  else                Cbf[t * 256 + ch - 512] = val;
}

// 5b) LDS-tiled bf16 transpose [kN][256] -> [256][kN]   <<<dim3(512,8), dim3(32,8)>>>
__global__ void k_transpose(const u16* __restrict__ in, u16* __restrict__ outT) {
  __shared__ u16 tile[32][33];
  int t0 = blockIdx.x * 32, d0 = blockIdx.y * 32;
  int tx = threadIdx.x, ty = threadIdx.y;
  for (int r = ty; r < 32; r += 8)
    tile[r][tx] = in[(t0 + r) * 256 + d0 + tx];
  __syncthreads();
  for (int r = ty; r < 32; r += 8)
    outT[(d0 + r) * kN + t0 + tx] = tile[tx][r];
}

// 6) dt = softplus(dt_raw + bias); cs = cumsum(dt*A) per chunk; chunk decay
//    <<<NCH*H, 64>>>
__global__ void k_dtcs(const float* __restrict__ zx, const float* __restrict__ dtb,
                       const float* __restrict__ Alog, float* __restrict__ dt,
                       float* __restrict__ cs, float* __restrict__ cdec) {
  __shared__ float arr[kCHUNK];
  int c = blockIdx.x >> 4, h = blockIdx.x & 15, j = threadIdx.x;
  int t = c * kCHUNK + j;
  float d = softplusf_(zx[t * kDPROJ + 1024 + h] + dtb[h]);
  float A = -__expf(Alog[h]);
  arr[j] = d * A;
  __syncthreads();
  // inclusive Hillis-Steele scan over 64 elements
  for (int off = 1; off < kCHUNK; off <<= 1) {
    float add = (j >= off) ? arr[j - off] : 0.f;
    __syncthreads();
    arr[j] += add;
    __syncthreads();
  }
  float csv = arr[j];
  dt[t * kH + h] = d;
  cs[t * kH + h] = csv;
  if (j == kCHUNK - 1) cdec[c * kH + h] = __expf(csv);
}

// 7) xw = x * exp(cs_last - cs) * dt   (weighted x for chunk-state GEMM)  <<<N,256>>>
__global__ void k_xw(const u16* __restrict__ xbf, const float* __restrict__ dt,
                     const float* __restrict__ cs, u16* __restrict__ xw) {
  int t = blockIdx.x, d = threadIdx.x, h = d >> 4;
  int c = t / kCHUNK;
  float last = cs[(c * kCHUNK + kCHUNK - 1) * kH + h];
  float w = __expf(last - cs[t * kH + h]) * dt[t * kH + h];
  xw[t * 256 + d] = f2bf(bf2f(xbf[t * 256 + d]) * w);
}

// 8) CB[c,i,j] = sum_n C[i,n]*B[j,n]   (64x64, K=256 per chunk)   <<<512, 256>>>
__global__ void k_cb(const u16* __restrict__ Cbf, const u16* __restrict__ Bbf,
                     float* __restrict__ CB) {
  int wid = threadIdx.x >> 5, lane = threadIdx.x & 31;
  int lo = lane & 15, hi = lane >> 4;
  int tile = blockIdx.x * 8 + wid;                 // 256 chunks * 16 tiles
  int c = tile >> 4, it = (tile >> 2) & 3, jt = tile & 3;
  const u16* Cc = Cbf + c * kCHUNK * kNS;
  const u16* Bc = Bbf + c * kCHUNK * kNS;
  v8f acc = {};
#pragma unroll
  for (int kk = 0; kk < kNS; kk += 32) {
    v16bf a = load_a(Cc, kNS, it * 16 + lo, kk, hi);
    v16bf b = load_b(Bc, kNS, jt * 16 + lo, kk, hi);   // Bm rows = B^T cols: contiguous
    acc = wmma_bf16(a, b, acc);
  }
#pragma unroll
  for (int v = 0; v < 8; ++v)
    CB[c * 4096 + (it * 16 + v + hi * 8) * 64 + jt * 16 + lo] = acc[v];
}

// 9) S[c,h,p,n] = sum_j xw[j,h,p]*B[j,n]   (16x256, K=64 per (c,h))   <<<8192, 256>>>
//    A from xwT [feat][token], B from BbfT [n][token]: both K-contiguous.
__global__ void k_states(const u16* __restrict__ xwT, const u16* __restrict__ BbfT,
                         float* __restrict__ S) {
  int wid = threadIdx.x >> 5, lane = threadIdx.x & 31;
  int lo = lane & 15, hi = lane >> 4;
  int tile = blockIdx.x * 8 + wid;                 // 256*16*16
  int c = tile >> 8, h = (tile >> 4) & 15, nt = tile & 15;
  const u16* Aw = xwT + c * kCHUNK;                // row = h*16+p, lda = kN
  const u16* Bc = BbfT + c * kCHUNK;               // col = n,      nst = kN
  v8f acc = {};
#pragma unroll
  for (int kk = 0; kk < kCHUNK; kk += 32) {
    v16bf a = load_a(Aw, kN, h * 16 + lo, kk, hi);
    v16bf b = load_b(Bc, kN, nt * 16 + lo, kk, hi);
    acc = wmma_bf16(a, b, acc);
  }
#pragma unroll
  for (int v = 0; v < 8; ++v)
    S[((c * kH + h) * kP + v + hi * 8) * kNS + nt * 16 + lo] = acc[v];
}

// 10) sequential inter-chunk scan over 256 chunks; emits bf16 entering-state
//     one thread per (h,p,n)      <<<256, 256>>>
__global__ void k_scan(const float* __restrict__ S, const float* __restrict__ cdec,
                       u16* __restrict__ prevb) {
  int flat = blockIdx.x * 256 + threadIdx.x;       // 65536
  int n = flat & 255, p = (flat >> 8) & 15, h = flat >> 12;
  float prev = 0.f;
  for (int c = 0; c < kNCH; ++c) {
    int idx = ((c * kH + h) * kP + p) * kNS + n;
    float s = S[idx];
    prevb[idx] = f2bf(prev);
    prev = cdec[c * kH + h] * prev + s;
  }
}

// 11) Ydiag: per (c,h) build M[i][j] = mask * CB * exp(cs_i - cs_j) * dt_j in LDS
//     while async-staging the x tile (16 x 64 bf16) into LDS, then
//     Y_h = M @ X_h  (64x16, K=64)     <<<NCH*H, 128>>>
__global__ void k_ydiag(const float* __restrict__ CB, const float* __restrict__ cs,
                        const float* __restrict__ dt, const u16* __restrict__ xbfT,
                        float* __restrict__ Y) {
  __shared__ u16 m[64 * 64];        // decay matrix, row-major [i][j]
  __shared__ u16 xs[16 * 64];       // x tile, [p][j]
  int c = blockIdx.x >> 4, h = blockIdx.x & 15;
  int tid = threadIdx.x;

  // async global->LDS copy of x tile: 16 rows x 128B; 128 lanes x 16B each
  {
    int r = tid >> 3, seg = tid & 7;
    const u16* gsrc = xbfT + (h * 16 + r) * kN + c * kCHUNK + seg * 8;
    u32 ldst = (u32)(size_t)(&xs[0]) + (u32)(r * 128 + seg * 16);
    u64 ga = (u64)(size_t)gsrc;
    asm volatile("global_load_async_to_lds_b128 %0, %1, off"
                 :: "v"(ldst), "v"(ga) : "memory");
  }

  // overlap: build decay matrix with VALU while the async copy is in flight
  for (int e = tid; e < 4096; e += 128) {
    int i = e >> 6, j = e & 63;
    float v = 0.f;
    if (i >= j)
      v = CB[c * 4096 + e] *
          __expf(cs[(c * kCHUNK + i) * kH + h] - cs[(c * kCHUNK + j) * kH + h]) *
          dt[(c * kCHUNK + j) * kH + h];
    m[e] = f2bf(v);
  }
  asm volatile("s_wait_asynccnt 0x0" ::: "memory");
  __syncthreads();

  int w = tid >> 5, lane = tid & 31;
  int lo = lane & 15, hi = lane >> 4;
  v8f acc = {};
#pragma unroll
  for (int kk = 0; kk < kCHUNK; kk += 32) {
    BFrag a;                                        // A from LDS m, lda = 64
    int abase = (w * 16 + lo) * 64 + kk + hi * 8;
    BFrag b;                                        // B from LDS xs, col=p, k=j
    int bbase = lo * 64 + kk + hi * 8;
#pragma unroll
    for (int i = 0; i < 8; ++i) { a.u[i] = m[abase + i]; a.u[8 + i] = m[abase + 16 + i]; }
#pragma unroll
    for (int i = 0; i < 8; ++i) { b.u[i] = xs[bbase + i]; b.u[8 + i] = xs[bbase + 16 + i]; }
    acc = wmma_bf16(a.v, b.v, acc);
  }
#pragma unroll
  for (int v = 0; v < 8; ++v)
    Y[(c * kCHUNK + w * 16 + v + hi * 8) * 256 + h * 16 + lo] = acc[v];
}

// 12) Yoff: Y[i,h,p] += exp(cs_i_h) * sum_n C[i,n]*prev[h,p,n]  (64x16, K=256)
//     <<<NCH*H, 128>>>
__global__ void k_yoff(const u16* __restrict__ Cbf, const u16* __restrict__ prevb,
                       const float* __restrict__ cs, float* __restrict__ Y) {
  int c = blockIdx.x >> 4, h = blockIdx.x & 15;
  int w = threadIdx.x >> 5, lane = threadIdx.x & 31;
  int lo = lane & 15, hi = lane >> 4;
  const u16* Cc = Cbf + c * kCHUNK * kNS;
  const u16* Pv = prevb + (c * kH + h) * kP * kNS;  // [p][n], n contiguous
  v8f acc = {};
#pragma unroll
  for (int kk = 0; kk < kNS; kk += 32) {
    v16bf a = load_a(Cc, kNS, w * 16 + lo, kk, hi);
    v16bf b = load_b(Pv, kNS, lo, kk, hi);          // col = p
    acc = wmma_bf16(a, b, acc);
  }
#pragma unroll
  for (int v = 0; v < 8; ++v) {
    int tok = c * kCHUNK + w * 16 + v + hi * 8;
    float sc = __expf(cs[tok * kH + h]);
    Y[tok * 256 + h * 16 + lo] += sc * acc[v];
  }
}

// 13) y = Y + D*x; g = y*silu(z); yg = rmsnorm(g)*gnorm_w -> bf16   <<<N, 256>>>
__global__ void k_gate(const float* __restrict__ Y, const u16* __restrict__ xbf,
                       const float* __restrict__ zx, const float* __restrict__ Dh,
                       const float* __restrict__ gw, u16* __restrict__ yg) {
  __shared__ float s[kDIN];
  int t = blockIdx.x, d = threadIdx.x, h = d >> 4;
  float yv = Y[t * 256 + d] + Dh[h] * bf2f(xbf[t * 256 + d]);
  float z = zx[t * kDPROJ + d];
  float g = yv * siluf_(z);
  s[d] = g * g;
  __syncthreads();
  for (int o = kDIN / 2; o > 0; o >>= 1) { if (d < o) s[d] += s[d + o]; __syncthreads(); }
  float inv = __frsqrt_rn(s[0] * (1.f / kDIN) + 1e-5f);
  yg[t * kDIN + d] = f2bf(g * inv * gw[d]);
}

// 14) hidden[N,128] = yg @ W_out  (M=16384,K=256,Nout=128)   <<<1024, 256>>>
__global__ void k_outproj(const u16* __restrict__ yg, const u16* __restrict__ Wt,
                          float* __restrict__ hidden) {
  int wid = threadIdx.x >> 5, lane = threadIdx.x & 31;
  int lo = lane & 15, hi = lane >> 4;
  int tile = blockIdx.x * 8 + wid;                 // 1024 * 8 tiles
  int mt = tile >> 3, nt = tile & 7;
  __builtin_prefetch(Wt + (nt * 16 + lo) * kDIN, 0, 1);
  v8f acc = {};
#pragma unroll
  for (int kk = 0; kk < kDIN; kk += 32) {
    v16bf a = load_a(yg, kDIN, mt * 16 + lo, kk, hi);
    v16bf b = load_b(Wt, kDIN, nt * 16 + lo, kk, hi);
    acc = wmma_bf16(a, b, acc);
  }
#pragma unroll
  for (int v = 0; v < 8; ++v)
    hidden[(mt * 16 + v + hi * 8) * kC + nt * 16 + lo] = acc[v];
}

// 15) residual += hidden; f2 = f + rmsnorm(residual)*normf_w    <<<N, 128>>>
__global__ void k_final(const float* __restrict__ hidden, float* __restrict__ residual,
                        const float* __restrict__ f, const float* __restrict__ nfw,
                        float* __restrict__ f2) {
  __shared__ float s[kC];
  int t = blockIdx.x, c = threadIdx.x;
  float r = hidden[t * kC + c] + residual[t * kC + c];
  s[c] = r * r;
  __syncthreads();
  for (int o = kC / 2; o > 0; o >>= 1) { if (c < o) s[c] += s[c + o]; __syncthreads(); }
  float inv = __frsqrt_rn(s[0] * (1.f / kC) + 1e-5f);
  f2[t * kC + c] = f[t * kC + c] + r * inv * nfw[c];
}

// 16) BatchNorm batch stats per channel     <<<128, 256>>>
__global__ void k_bnstats(const float* __restrict__ f2, float* __restrict__ stats) {
  __shared__ float s1[256], s2[256];
  int ch = blockIdx.x, tid = threadIdx.x;
  float a = 0.f, b = 0.f;
  for (int r = tid; r < kN; r += 256) {
    float v = f2[r * kC + ch];
    a += v; b += v * v;
  }
  s1[tid] = a; s2[tid] = b;
  __syncthreads();
  for (int o = 128; o > 0; o >>= 1) {
    if (tid < o) { s1[tid] += s1[tid + o]; s2[tid] += s2[tid + o]; }
    __syncthreads();
  }
  if (tid == 0) {
    float mu = s1[0] * (1.f / kN);
    stats[ch] = mu;
    stats[kC + ch] = s2[0] * (1.f / kN) - mu * mu;
  }
}

// 17) BatchNorm apply      <<<N*C/256, 256>>>
__global__ void k_bnapply(const float* __restrict__ f2, const float* __restrict__ stats,
                          const float* __restrict__ gamma, const float* __restrict__ beta,
                          float* __restrict__ out) {
  int idx = blockIdx.x * 256 + threadIdx.x;
  int ch = idx & (kC - 1);
  float mu = stats[ch], var = stats[kC + ch];
  out[idx] = (f2[idx] - mu) * __frsqrt_rn(var + 1e-5f) * gamma[ch] + beta[ch];
}

// ---------------------------------------------------------------------------
extern "C" void kernel_launch(void* const* d_in, const int* in_sizes, int n_in,
                              void* d_out, int out_size, void* d_ws, size_t ws_size,
                              hipStream_t stream) {
  const float* f      = (const float*)d_in[0];
  const float* f_gs   = (const float*)d_in[1];
  const float* pos    = (const float*)d_in[2];
  const float* W_in   = (const float*)d_in[3];   // [2,128,1040]
  const float* convw  = (const float*)d_in[4];   // [2,768,4]
  const float* convb  = (const float*)d_in[5];   // [2,768]
  const float* dtb    = (const float*)d_in[6];   // [2,16]
  const float* Alog   = (const float*)d_in[7];   // [2,16]
  const float* Dh     = (const float*)d_in[8];   // [2,16]
  const float* gw     = (const float*)d_in[9];   // [2,256]
  const float* W_out  = (const float*)d_in[10];  // [2,256,128]
  const float* lnw    = (const float*)d_in[11];  // [2,128]
  const float* normfw = (const float*)d_in[12];
  const float* gamma  = (const float*)d_in[13];
  const float* beta   = (const float*)d_in[14];
  float* out = (float*)d_out;

  char* ws = (char*)d_ws;
  size_t off = 0;
  auto take = [&](size_t bytes) -> char* {
    char* p = ws + off;
    off += (bytes + 255) & ~(size_t)255;
    return p;
  };
  float* hidden   = (float*)take((size_t)kN * kC * 4);
  float* residual = (float*)take((size_t)kN * kC * 4);
  float* f2       = (float*)take((size_t)kN * kC * 4);
  u16*   xnbf     = (u16*)  take((size_t)kN * kC * 2);
  u16*   WtA      = (u16*)  take((size_t)kDPROJ * kC * 2);
  u16*   WtO      = (u16*)  take((size_t)kC * kDIN * 2);
  float* zx       = (float*)take((size_t)kN * kDPROJ * 4);
  u16*   xbf      = (u16*)  take((size_t)kN * 256 * 2);
  u16*   Bbf      = (u16*)  take((size_t)kN * 256 * 2);
  u16*   Cbf      = (u16*)  take((size_t)kN * 256 * 2);
  u16*   xwbf     = (u16*)  take((size_t)kN * 256 * 2);
  u16*   xbfT     = (u16*)  take((size_t)kN * 256 * 2);
  u16*   BbfT     = (u16*)  take((size_t)kN * 256 * 2);
  u16*   xwT      = (u16*)  take((size_t)kN * 256 * 2);
  float* dt       = (float*)take((size_t)kN * kH * 4);
  float* cs       = (float*)take((size_t)kN * kH * 4);
  float* cdec     = (float*)take((size_t)kNCH * kH * 4);
  float* CB       = (float*)take((size_t)kNCH * 4096 * 4);
  float* S        = (float*)take((size_t)kNCH * kH * kP * kNS * 4);
  u16*   prevb    = (u16*)  take((size_t)kNCH * kH * kP * kNS * 2);
  float* Y        = (float*)take((size_t)kN * 256 * 4);
  u16*   ygbf     = (u16*)  take((size_t)kN * kDIN * 2);
  float* stats    = (float*)take((size_t)2 * kC * 4);

  dim3 trGrid(kN / 32, 256 / 32);     // 512 x 8
  dim3 trBlk(32, 8);

  k_embed<<<kN, 128, 0, stream>>>(f, f_gs, pos, hidden);

  for (int l = 0; l < 2; ++l) {
    const float* Wi  = W_in  + (size_t)l * kC * kDPROJ;
    const float* cw  = convw + (size_t)l * kCONV * 4;
    const float* cb  = convb + (size_t)l * kCONV;
    const float* db  = dtb   + (size_t)l * kH;
    const float* Al  = Alog  + (size_t)l * kH;
    const float* Dl  = Dh    + (size_t)l * kH;
    const float* gwl = gw    + (size_t)l * kDIN;
    const float* Wo  = W_out + (size_t)l * kDIN * kC;
    const float* lnl = lnw   + (size_t)l * kC;

    k_prep_rms<<<kN, 128, 0, stream>>>(hidden, residual, lnl, xnbf, l == 0 ? 1 : 0);
    k_wt_in  <<<(kDPROJ * kC) / 256, 256, 0, stream>>>(Wi, WtA);
    k_wt_out <<<(kC * kDIN) / 256, 256, 0, stream>>>(Wo, WtO);
    k_inproj <<<(kN / 16) * 65 / 8, 256, 0, stream>>>(xnbf, WtA, zx);        // 8320
    k_conv   <<<(kN * kCONV) / 256, 256, 0, stream>>>(zx, cw, cb, xbf, Bbf, Cbf);
    k_transpose<<<trGrid, trBlk, 0, stream>>>(xbf, xbfT);
    k_transpose<<<trGrid, trBlk, 0, stream>>>(Bbf, BbfT);
    k_dtcs   <<<kNCH * kH, 64, 0, stream>>>(zx, db, Al, dt, cs, cdec);
    k_xw     <<<kN, 256, 0, stream>>>(xbf, dt, cs, xwbf);
    k_transpose<<<trGrid, trBlk, 0, stream>>>(xwbf, xwT);
    k_cb     <<<kNCH * 16 / 8, 256, 0, stream>>>(Cbf, Bbf, CB);              // 512
    k_states <<<kNCH * kH * 16 / 8, 256, 0, stream>>>(xwT, BbfT, S);         // 8192
    k_scan   <<<(kH * kP * kNS) / 256, 256, 0, stream>>>(S, cdec, prevb);    // 256
    k_ydiag  <<<kNCH * kH, 128, 0, stream>>>(CB, cs, dt, xbfT, Y);
    k_yoff   <<<kNCH * kH, 128, 0, stream>>>(Cbf, prevb, cs, Y);
    k_gate   <<<kN, 256, 0, stream>>>(Y, xbf, zx, Dl, gwl, ygbf);
    k_outproj<<<(kN / 16) * 8 / 8, 256, 0, stream>>>(ygbf, WtO, hidden);     // 1024
  }

  k_final  <<<kN, 128, 0, stream>>>(hidden, residual, f, normfw, f2);
  k_bnstats<<<kC, 256, 0, stream>>>(f2, stats);
  k_bnapply<<<(kN * kC) / 256, 256, 0, stream>>>(f2, stats, gamma, beta, out);
}
